// DioraBase_56229711839540
// MI455X (gfx1250) — matrix operand order repaired and checked by reading, not compiled, not run on hardware
//
#include <hip/hip_runtime.h>
#include <hip/hip_bf16.h>

// ---------------------------------------------------------------------------
// DIORA inside pass for MI455X (gfx1250): bf16 WMMA for all 400x400 matmuls.
// Weights pre-shuffled into WMMA B-fragment order (2x global_load_b128 per
// fragment); A operands gathered per-lane with aligned b128 loads.
// K padded 400->416 (13 chunks of 32). Epilogues use v_tanh_f32 / v_exp_f32.
// ---------------------------------------------------------------------------

typedef __attribute__((ext_vector_type(16))) __bf16 bf16x16;
typedef __attribute__((ext_vector_type(8)))  float  f32x8;

union BFrag { bf16x16 v; unsigned int u[8]; };

#define DEVFN static __device__ __forceinline__

constexpr int    BB   = 128;
constexpr int    TT   = 32;
constexpr int    DD   = 400;
constexpr int    DP   = 416;              // padded K/N
constexpr int    NC   = 528;              // T*(T+1)/2 cells
constexpr int    KCH  = 13;               // DP / 32
constexpr int    CT   = DP / 16;          // 26 column tiles
constexpr size_t MMAX = (size_t)BB * 16 * 16;   // max rows of a level GEMM

DEVFN unsigned short f2bfu(float f){
  unsigned int u = __builtin_bit_cast(unsigned int, f);
  return (unsigned short)((u + 0x7FFFu + ((u >> 16) & 1u)) >> 16);
}
DEVFN float bf2f(unsigned short s){
  unsigned int u = ((unsigned int)s) << 16;
  return __builtin_bit_cast(float, u);
}
DEVFN int coffs(int k){ return k * TT - (k * (k - 1)) / 2; }

// gfx1250 has a hardware TANH transcendental (TRANS32; co-executes with WMMA).
DEVFN float fast_tanh(float x){
#if __has_builtin(__builtin_amdgcn_tanhf)
  return __builtin_amdgcn_tanhf(x);
#else
  float y;
  asm volatile("v_tanh_f32 %0, %1\n\tv_nop" : "=v"(y) : "v"(x));
  return y;
#endif
}

// A fragment (16x32 bf16) loaded straight from a (gathered) row-major row:
// lanes 0-15 = rows; v0..3 <-> K pairs at kBase+hi*8, v4..7 at kBase+16+hi*8.
// Both 16B chunks are 16-byte aligned (row stride 832 B, kBase*2 % 16 == 0).
DEVFN void loadAg(const unsigned short* __restrict__ rowp, int kBase, int hi, BFrag& f){
  const uint4 a = *(const uint4*)(rowp + kBase + hi * 8);
  const uint4 b = *(const uint4*)(rowp + kBase + 16 + hi * 8);
  f.u[0] = a.x; f.u[1] = a.y; f.u[2] = a.z; f.u[3] = a.w;
  f.u[4] = b.x; f.u[5] = b.y; f.u[6] = b.z; f.u[7] = b.w;
}
// B fragment from fragment-ordered weights: 8 consecutive dwords per lane.
DEVFN void loadBf(const unsigned int* __restrict__ W, int ct, int kc, int lane, BFrag& f){
  const uint4* p = (const uint4*)(W + (((size_t)ct * KCH + kc) * 32 + lane) * 8);
  const uint4 a = p[0], b = p[1];
  f.u[0] = a.x; f.u[1] = a.y; f.u[2] = a.z; f.u[3] = a.w;
  f.u[4] = b.x; f.u[5] = b.y; f.u[6] = b.z; f.u[7] = b.w;
}
DEVFN f32x8 bwmma(const BFrag& a, const BFrag& b, f32x8 c){
  return __builtin_amdgcn_wmma_f32_16x16x32_bf16(false, a.v, false, b.v,
                                                 (short)0, c, false, false);
}

// ---------------------------------------------------------------------------
// Setup kernels
// ---------------------------------------------------------------------------
// Shuffle a 400x400 f32 (K x N, row-major) weight into B-fragment order:
// dst[((ct*KCH + kc)*32 + lane)*8 + v] = pack(W[k][n], W[k+1][n])
//   n = ct*16 + (lane&15); k = kc*32 + (v>>2)*16 + (lane>>4)*8 + 2*(v&3); zeros past 400.
__global__ void k_shuffle_w(const float* __restrict__ src, unsigned int* __restrict__ dst){
  int idx = blockIdx.x * 256 + threadIdx.x;
  if (idx >= CT * KCH * 256) return;
  int v    = idx & 7;
  int lane = (idx >> 3) & 31;
  int kc   = (idx >> 8) % KCH;
  int ct   = (idx >> 8) / KCH;
  int n  = ct * 16 + (lane & 15);
  int k  = kc * 32 + (v >> 2) * 16 + (lane >> 4) * 8 + 2 * (v & 3);
  float f0 = (k     < DD && n < DD) ? src[(size_t)k       * DD + n] : 0.f;
  float f1 = (k + 1 < DD && n < DD) ? src[(size_t)(k + 1) * DD + n] : 0.f;
  dst[idx] = (unsigned int)f2bfu(f0) | ((unsigned int)f2bfu(f1) << 16);
}

__global__ void k_convert_x(const float* __restrict__ src, unsigned short* __restrict__ dst){
  size_t idx = (size_t)blockIdx.x * 256 + threadIdx.x;
  if (idx >= (size_t)BB * TT * DP) return;
  size_t r = idx / DP; int c = (int)(idx % DP);
  float v = (c < DD) ? src[r * DD + c] : 0.f;
  dst[idx] = f2bfu(v);
}

__global__ void k_zero(float* __restrict__ p, int n){
  int i = blockIdx.x * 256 + threadIdx.x;
  if (i < n) p[i] = 0.f;
}

// ---------------------------------------------------------------------------
// Leaf GEMM: hbuf = tanh(xb @ Wh + bh), cbuf = tanh(xb @ Wc + bc)   (bf16 out)
// 1 wave per block, 2 row-tiles (32 rows) x 1 col-tile; grid = (rows/32, 26)
// ---------------------------------------------------------------------------
__global__ void __launch_bounds__(32)
k_leaf_gemm(const unsigned short* __restrict__ xb,
            const unsigned int* __restrict__ Wh,
            const unsigned int* __restrict__ Wc,
            const float* __restrict__ bh, const float* __restrict__ bc,
            unsigned short* __restrict__ hbuf, unsigned short* __restrict__ cbuf){
  const int lane  = threadIdx.x;
  const int laneL = lane & 15, hi = lane >> 4;
  const size_t rowBase = (size_t)blockIdx.x * 32;
  const int ct = blockIdx.y;
  const int n  = ct * 16 + laneL;

  const unsigned short* arow[2];
#pragma unroll
  for (int rt = 0; rt < 2; ++rt)
    arow[rt] = xb + (rowBase + rt * 16 + laneL) * DP;

  f32x8 accH[2] = {{}, {}};
  f32x8 accC[2] = {{}, {}};
#pragma unroll 1
  for (int kc = 0; kc < KCH; ++kc){
    const int kBase = kc * 32;
    __builtin_prefetch(Wh + (((size_t)ct * KCH + kc + 1) * 32 + lane) * 8, 0, 3);
    BFrag fh, fc;
    loadBf(Wh, ct, kc, lane, fh);
    loadBf(Wc, ct, kc, lane, fc);
#pragma unroll
    for (int rt = 0; rt < 2; ++rt){
      BFrag a;
      loadAg(arow[rt], kBase, hi, a);
      accH[rt] = bwmma(a, fh, accH[rt]);
      accC[rt] = bwmma(a, fc, accC[rt]);
    }
  }

  const float inb  = (n < DD) ? 1.f : 0.f;
  const int   nclamp = (n < DD) ? n : 0;
  const float bhv = bh[nclamp] * inb;
  const float bcv = bc[nclamp] * inb;
#pragma unroll
  for (int rt = 0; rt < 2; ++rt){
#pragma unroll
    for (int r = 0; r < 8; ++r){
      size_t row = rowBase + rt * 16 + r + hi * 8;
      hbuf[row * DP + n] = f2bfu(fast_tanh(accH[rt][r] + bhv));
      cbuf[row * DP + n] = f2bfu(fast_tanh(accC[rt][r] + bcv));
    }
  }
}

// ---------------------------------------------------------------------------
// Leaf normalize: unit-norm rows of hbuf/cbuf into d_out (f32) + bf16 mirrors
// grid = B*T, block = 256
// ---------------------------------------------------------------------------
__global__ void __launch_bounds__(256)
k_leaf_norm(const unsigned short* __restrict__ hbuf,
            const unsigned short* __restrict__ cbuf,
            float* __restrict__ chout,
            unsigned short* __restrict__ chb, unsigned short* __restrict__ ccb,
            float* __restrict__ cs){
  __shared__ float agg[DP];
  __shared__ float red[256];
  const int tid = threadIdx.x;
  const int row = blockIdx.x;              // b*T + t
  const int b = row >> 5, t = row & 31;
  const size_t orow = (size_t)b * NC + t;  // leaf cell index == t

  float ss = 0.f;
  for (int d = tid; d < DP; d += 256){
    float v = bf2f(hbuf[(size_t)row * DP + d]);
    agg[d] = v; ss += v * v;
  }
  red[tid] = ss; __syncthreads();
  for (int s = 128; s > 0; s >>= 1){ if (tid < s) red[tid] += red[tid + s]; __syncthreads(); }
  float rn = 1.f / fmaxf(sqrtf(red[0]), 1e-8f);
  for (int d = tid; d < DP; d += 256){
    float v = agg[d] * rn;
    if (d < DD) chout[orow * DD + d] = v;
    chb[orow * DP + d] = f2bfu(v);
  }
  __syncthreads();
  ss = 0.f;
  for (int d = tid; d < DP; d += 256){
    float v = bf2f(cbuf[(size_t)row * DP + d]);
    agg[d] = v; ss += v * v;
  }
  red[tid] = ss; __syncthreads();
  for (int s = 128; s > 0; s >>= 1){ if (tid < s) red[tid] += red[tid + s]; __syncthreads(); }
  rn = 1.f / fmaxf(sqrtf(red[0]), 1e-8f);
  for (int d = tid; d < DP; d += 256) ccb[orow * DP + d] = f2bfu(agg[d] * rn);
  if (tid == 0) cs[orow] = 0.f;
}

// ---------------------------------------------------------------------------
// Per-level pair GEMM. Rows r = ((b*L + i)*N + k). Per-lane gathered A rows,
// fragment-ordered B. 1 wave per block, 2 row-tiles x 1 col-tile,
// 10 wmma per K-chunk:
//   accH = lh@Ulh + rh@Urh ; accC = lc@Ulc + rc@Urc ; accS = lh@Wscore
// Epilogue: tanh(+bias) -> bf16 hbuf/cbuf; bilinear score folded via
// LDS ds_add_f32 + one global atomic per row into sbuf.
// grid = (M/32, 26), block = 32
// ---------------------------------------------------------------------------
__global__ void __launch_bounds__(32)
k_pair_gemm(const unsigned short* __restrict__ chb,
            const unsigned short* __restrict__ ccb,
            const unsigned int* __restrict__ wULH,
            const unsigned int* __restrict__ wURH,
            const unsigned int* __restrict__ wULC,
            const unsigned int* __restrict__ wURC,
            const unsigned int* __restrict__ wSC,
            const float* __restrict__ b_h, const float* __restrict__ b_c,
            unsigned short* __restrict__ hbuf,
            unsigned short* __restrict__ cbuf,
            float* __restrict__ sbuf,
            int level, int L, int N){
  __shared__ unsigned long long rofsS[32];
  __shared__ float sred[32];

  const int lane  = threadIdx.x;
  const int laneL = lane & 15, hi = lane >> 4;
  const size_t rowBase = (size_t)blockIdx.x * 32;
  const int ct = blockIdx.y;
  const int n  = ct * 16 + laneL;

  sred[lane] = 0.f;

  unsigned long long lofs[2], rofs[2];
#pragma unroll
  for (int rt = 0; rt < 2; ++rt){
    long long r = (long long)rowBase + rt * 16 + laneL;
    int LN = L * N;
    int b  = (int)(r / LN);
    int pr = (int)(r % LN);
    int i  = pr / N;
    int k  = pr % N;
    int lcell = coffs(k) + i;
    int rcell = coffs(level - k - 1) + i + k + 1;
    lofs[rt] = ((unsigned long long)b * NC + lcell) * DP;
    rofs[rt] = ((unsigned long long)b * NC + rcell) * DP;
    if (hi == 0) rofsS[rt * 16 + laneL] = rofs[rt];
  }
  __syncthreads();

  f32x8 accH[2] = {{}, {}};
  f32x8 accC[2] = {{}, {}};
  f32x8 accS[2] = {{}, {}};
#pragma unroll 1
  for (int kc = 0; kc < KCH; ++kc){
    const int kBase = kc * 32;
    __builtin_prefetch(wULH + (((size_t)ct * KCH + kc + 1) * 32 + lane) * 8, 0, 3);
    BFrag fLH, fRH, fLC, fRC, fSC;
    loadBf(wULH, ct, kc, lane, fLH);
    loadBf(wURH, ct, kc, lane, fRH);
    loadBf(wULC, ct, kc, lane, fLC);
    loadBf(wURC, ct, kc, lane, fRC);
    loadBf(wSC,  ct, kc, lane, fSC);
#pragma unroll
    for (int rt = 0; rt < 2; ++rt){
      BFrag aLH, aRH, aLC, aRC;
      loadAg(chb + lofs[rt], kBase, hi, aLH);
      loadAg(chb + rofs[rt], kBase, hi, aRH);
      loadAg(ccb + lofs[rt], kBase, hi, aLC);
      loadAg(ccb + rofs[rt], kBase, hi, aRC);
      accH[rt] = bwmma(aLH, fLH, accH[rt]);
      accH[rt] = bwmma(aRH, fRH, accH[rt]);
      accC[rt] = bwmma(aLC, fLC, accC[rt]);
      accC[rt] = bwmma(aRC, fRC, accC[rt]);
      accS[rt] = bwmma(aLH, fSC, accS[rt]);
    }
  }

  const float inb  = (n < DD) ? 1.f : 0.f;
  const int   nclamp = (n < DD) ? n : 0;
  const float bhv = b_h[nclamp] * inb;
  const float bcv = b_c[nclamp] * inb;
#pragma unroll
  for (int rt = 0; rt < 2; ++rt){
#pragma unroll
    for (int r = 0; r < 8; ++r){
      int m = rt * 16 + r + hi * 8;
      size_t row = rowBase + m;
      hbuf[row * DP + n] = f2bfu(fast_tanh(accH[rt][r] + bhv));
      cbuf[row * DP + n] = f2bfu(fast_tanh(accC[rt][r] + bcv));
      float rv = bf2f(chb[rofsS[m] + n]);      // rh value, same column
      atomicAdd(&sred[m], accS[rt][r] * rv);   // ds_add_f32
    }
  }
  __syncthreads();
  atomicAdd(&sbuf[rowBase + lane], sred[lane]);
}

// ---------------------------------------------------------------------------
// Per-level combine: scores + softmax over N splits, aggregate, unit-norm.
// grid = B*L, block = 256
// ---------------------------------------------------------------------------
__global__ void __launch_bounds__(256)
k_combine(const unsigned short* __restrict__ hbuf,
          const unsigned short* __restrict__ cbuf,
          const float* __restrict__ sbuf,
          float* __restrict__ cs,
          float* __restrict__ chout,
          unsigned short* __restrict__ chb, unsigned short* __restrict__ ccb,
          int level, int L, int N){
  __shared__ float sarr[32], parr[32];
  __shared__ float agg[DP];
  __shared__ float red[256];
  const int tid = threadIdx.x, blk = blockIdx.x;
  const int b = blk / L, i = blk % L;
  const size_t base = (size_t)blk * N;
  const int cell = coffs(level) + i;

  if (tid < N){
    int k = tid;
    int lcell = coffs(k) + i;
    int rcell = coffs(level - k - 1) + i + k + 1;
    sarr[k] = sbuf[base + k] + cs[(size_t)b * NC + lcell] + cs[(size_t)b * NC + rcell];
  }
  __syncthreads();
  if (tid == 0){
    float mx = -3.0e38f;
    for (int k = 0; k < N; ++k) mx = fmaxf(mx, sarr[k]);
    float sum = 0.f;
    for (int k = 0; k < N; ++k){ float e = __expf(sarr[k] - mx); parr[k] = e; sum += e; }
    float inv = 1.f / sum, sagg = 0.f;
    for (int k = 0; k < N; ++k){ parr[k] *= inv; sagg += parr[k] * sarr[k]; }
    cs[(size_t)b * NC + cell] = sagg;
  }
  __syncthreads();

  const size_t orow = (size_t)b * NC + cell;
  float ss = 0.f;
  for (int d = tid; d < DP; d += 256){
    float a = 0.f;
    for (int k = 0; k < N; ++k) a += parr[k] * bf2f(hbuf[(base + k) * DP + d]);
    agg[d] = a; ss += a * a;
  }
  red[tid] = ss; __syncthreads();
  for (int s = 128; s > 0; s >>= 1){ if (tid < s) red[tid] += red[tid + s]; __syncthreads(); }
  float rn = 1.f / fmaxf(sqrtf(red[0]), 1e-8f);
  for (int d = tid; d < DP; d += 256){
    float v = agg[d] * rn;
    if (d < DD) chout[orow * DD + d] = v;
    chb[orow * DP + d] = f2bfu(v);
  }
  __syncthreads();
  ss = 0.f;
  for (int d = tid; d < DP; d += 256){
    float a = 0.f;
    for (int k = 0; k < N; ++k) a += parr[k] * bf2f(cbuf[(base + k) * DP + d]);
    agg[d] = a; ss += a * a;
  }
  red[tid] = ss; __syncthreads();
  for (int s = 128; s > 0; s >>= 1){ if (tid < s) red[tid] += red[tid + s]; __syncthreads(); }
  rn = 1.f / fmaxf(sqrtf(red[0]), 1e-8f);
  for (int d = tid; d < DP; d += 256) ccb[orow * DP + d] = f2bfu(agg[d] * rn);
}

// ---------------------------------------------------------------------------
// Host launcher
// ---------------------------------------------------------------------------
extern "C" void kernel_launch(void* const* d_in, const int* in_sizes, int n_in,
                              void* d_out, int out_size, void* d_ws, size_t ws_size,
                              hipStream_t stream){
  (void)in_sizes; (void)n_in; (void)out_size; (void)ws_size;
  const float* x        = (const float*)d_in[0];
  const float* w_leaf_h = (const float*)d_in[1];
  const float* b_leaf_h = (const float*)d_in[2];
  const float* w_leaf_c = (const float*)d_in[3];
  const float* b_leaf_c = (const float*)d_in[4];
  const float* u_lh     = (const float*)d_in[5];
  const float* u_rh     = (const float*)d_in[6];
  const float* b_h      = (const float*)d_in[7];
  const float* u_lc     = (const float*)d_in[8];
  const float* u_rc     = (const float*)d_in[9];
  const float* b_c      = (const float*)d_in[10];
  const float* w_score  = (const float*)d_in[11];
  float* chout = (float*)d_out;

  // Workspace carve (~174 MB total)
  char* p = (char*)d_ws;
  auto carve = [&](size_t bytes) -> void* {
    void* r = (void*)p; p += (bytes + 255) & ~(size_t)255; return r;
  };
  const size_t cellsE = (size_t)BB * NC * DP;            // bf16 elements
  const size_t wElems = (size_t)CT * KCH * 256;          // u32 per weight
  unsigned short* chb  = (unsigned short*)carve(cellsE * 2);
  unsigned short* ccb  = (unsigned short*)carve(cellsE * 2);
  float*          cs   = (float*)carve((size_t)BB * NC * 4);
  unsigned short* hbuf = (unsigned short*)carve(MMAX * DP * 2);
  unsigned short* cbuf = (unsigned short*)carve(MMAX * DP * 2);
  float*          sbuf = (float*)carve(MMAX * 4);
  unsigned short* xb   = (unsigned short*)carve((size_t)BB * TT * DP * 2);
  unsigned int*   wbLH = (unsigned int*)carve(wElems * 4);
  unsigned int*   wbLC = (unsigned int*)carve(wElems * 4);
  unsigned int*   wULH = (unsigned int*)carve(wElems * 4);
  unsigned int*   wURH = (unsigned int*)carve(wElems * 4);
  unsigned int*   wULC = (unsigned int*)carve(wElems * 4);
  unsigned int*   wURC = (unsigned int*)carve(wElems * 4);
  unsigned int*   wSC  = (unsigned int*)carve(wElems * 4);

  const int wblocks = (int)((wElems + 255) / 256);
  k_shuffle_w<<<wblocks, 256, 0, stream>>>(w_leaf_h, wbLH);
  k_shuffle_w<<<wblocks, 256, 0, stream>>>(w_leaf_c, wbLC);
  k_shuffle_w<<<wblocks, 256, 0, stream>>>(u_lh, wULH);
  k_shuffle_w<<<wblocks, 256, 0, stream>>>(u_rh, wURH);
  k_shuffle_w<<<wblocks, 256, 0, stream>>>(u_lc, wULC);
  k_shuffle_w<<<wblocks, 256, 0, stream>>>(u_rc, wURC);
  k_shuffle_w<<<wblocks, 256, 0, stream>>>(w_score, wSC);

  const size_t xElems = (size_t)BB * TT * DP;
  k_convert_x<<<(unsigned)((xElems + 255) / 256), 256, 0, stream>>>(x, xb);

  // Leaves: (B*T) x 400 GEMMs -> tanh -> unit-norm cells 0..T-1
  const int leafRows = BB * TT;                 // 4096
  k_leaf_gemm<<<dim3(leafRows / 32, CT), 32, 0, stream>>>(
      xb, wbLH, wbLC, b_leaf_h, b_leaf_c, hbuf, cbuf);
  k_leaf_norm<<<leafRows, 256, 0, stream>>>(hbuf, cbuf, chout, chb, ccb, cs);

  // Levels
  for (int level = 1; level < TT; ++level){
    const int L = TT - level, N = level;
    const int M = BB * L * N;                   // rows (multiple of 32)
    k_zero<<<(M + 255) / 256, 256, 0, stream>>>(sbuf, M);
    k_pair_gemm<<<dim3(M / 32, CT), 32, 0, stream>>>(
        chb, ccb, wULH, wURH, wULC, wURC, wSC, b_h, b_c,
        hbuf, cbuf, sbuf, level, L, N);
    k_combine<<<BB * L, 256, 0, stream>>>(
        hbuf, cbuf, sbuf, cs, chout, chb, ccb, level, L, N);
  }
}
